// DCL_16449724745480
// MI455X (gfx1250) — compile-verified, matrix operand order checked by reading, not run on hardware
//
#include <hip/hip_runtime.h>

typedef __attribute__((ext_vector_type(8)))  unsigned short u16x8;
typedef __attribute__((ext_vector_type(16))) unsigned short u16x16;
typedef __attribute__((ext_vector_type(16))) __bf16         v16bf;
typedef __attribute__((ext_vector_type(8)))  float          v8f;

#define DIMS   256
#define CHUNKS 8          // 256 / 32 K-steps per WMMA
#define EPS    1e-6f

static __device__ __forceinline__ unsigned short f2bf(float f) {
    unsigned int u = __float_as_uint(f);
    u += 0x7FFFu + ((u >> 16) & 1u);      // round-to-nearest-even
    return (unsigned short)(u >> 16);
}

// Build one lane's v16bf WMMA operand fragment from a bf16 row.
// Per CDNA5 ISA 16-bit A/B layout: a lane's 16 values are two contiguous
// 8-element (16-byte) runs, 16 elements apart: [base..base+7] and [base+16..base+23].
static __device__ __forceinline__ v16bf load_bf16_frag(const unsigned short* p) {
    u16x8 lo = *(const u16x8*)(p);
    u16x8 hi = *(const u16x8*)(p + 16);
    u16x16 t = __builtin_shufflevector(lo, hi, 0,1,2,3,4,5,6,7,8,9,10,11,12,13,14,15);
    return __builtin_bit_cast(v16bf, t);
}

// ---------------- Phase A1: centers (fp32 mean of 8 rows), bf16 copy, ||c||^2 ----
__global__ __launch_bounds__(256) void dcl_centers(const float* __restrict__ inputs,
                                                   unsigned short* __restrict__ centers_bf,
                                                   float* __restrict__ norm_c, int N) {
    const int id = blockIdx.x;
    const int d  = threadIdx.x;
    const int halfN = N >> 1;
    float s = 0.f;
    #pragma unroll
    for (int p = 0; p < 4; ++p) {
        s += inputs[(size_t)(id * 4 + p) * DIMS + d];
        s += inputs[(size_t)(halfN + id * 4 + p) * DIMS + d];
    }
    const float c = s * 0.125f;
    centers_bf[(size_t)id * DIMS + d] = f2bf(c);
    __shared__ float red[256];
    red[d] = c * c;
    __syncthreads();
    for (int off = 128; off; off >>= 1) {
        if (d < off) red[d] += red[d + off];
        __syncthreads();
    }
    if (d == 0) norm_c[id] = red[0];
}

// ---------------- Phase A2: bf16 copy of inputs + ||x||^2 ------------------------
__global__ __launch_bounds__(256) void dcl_rows(const float* __restrict__ inputs,
                                                unsigned short* __restrict__ inputs_bf,
                                                float* __restrict__ norm_x) {
    const int row = blockIdx.x;
    const int d   = threadIdx.x;
    const float v = inputs[(size_t)row * DIMS + d];
    inputs_bf[(size_t)row * DIMS + d] = f2bf(v);
    __shared__ float red[256];
    red[d] = v * v;
    __syncthreads();
    for (int off = 128; off; off >>= 1) {
        if (d < off) red[d] += red[d + off];
        __syncthreads();
    }
    if (d == 0) norm_x[row] = red[0];
}

// ---------------- Phase B: fused WMMA GEMM + distance + mining (two passes) ------
// Each block owns 32 center rows (two 16-row A fragments resident in 128 VGPRs per
// wave); 8 waves split the column tiles. Two WMMAs per B fragment -> 32 FLOP per
// byte loaded from L2. Stats epilogue is fully branchless (cmp/cndmask/fma) so it
// co-issues under the WMMA shadow.
__global__ __launch_bounds__(256, 1) void dcl_main(
    const unsigned short* __restrict__ inputs_bf,
    const unsigned short* __restrict__ centers_bf,
    const float* __restrict__ norm_x,
    const float* __restrict__ norm_c,
    const int*   __restrict__ targets,
    float* __restrict__ row_an,
    float* __restrict__ pos_part,
    int N)
{
    const int tid  = threadIdx.x;
    const int wave = tid >> 5;
    const int lane = tid & 31;
    const int half = lane >> 4;         // which 16-lane half of the wave
    const int hl   = lane & 15;
    const int i0   = blockIdx.x << 5;   // 32 center rows per block

    // A operands: two 16x256 center tiles resident in VGPRs for the whole kernel.
    v16bf A0[CHUNKS], A1[CHUNKS];
    {
        const unsigned short* a0 = centers_bf + (size_t)(i0 + hl) * DIMS + half * 8;
        const unsigned short* a1 = a0 + (size_t)16 * DIMS;
        #pragma unroll
        for (int c = 0; c < CHUNKS; ++c) {
            A0[c] = load_bf16_frag(a0 + c * 32);
            A1[c] = load_bf16_frag(a1 + c * 32);
        }
    }
    // lrow(r): row-within-block for accumulator element r (r<8 -> acc0, else acc1).
    float nc[16]; int cid[16];
    #pragma unroll
    for (int r = 0; r < 16; ++r) {
        const int lrow = (r < 8) ? (half * 8 + r) : (16 + half * 8 + (r - 8));
        const int row  = i0 + lrow;
        nc[r]  = norm_c[row];
        cid[r] = targets[(size_t)row * 4];   // center_ids = targets[i*NUM_POS]
    }

    __shared__ float s_sum[8][32];
    __shared__ float s_cnt[8][32];
    __shared__ float s_pos[8][2];
    __shared__ float s_dneg[32];

    const int ntiles = N >> 4;

    // ---------------- pass 1: negative sums/counts + positive stats ------------
    float negsum[16], negcnt[16];
    #pragma unroll
    for (int r = 0; r < 16; ++r) { negsum[r] = 0.f; negcnt[r] = 0.f; }
    float possum = 0.f, poscnt = 0.f;

    for (int t = wave; t < ntiles; t += 8) {
        const int j0 = t << 4;
        const unsigned short* bb = inputs_bf + (size_t)(j0 + hl) * DIMS + half * 8;
        __builtin_prefetch(bb + (size_t)128 * DIMS, 0, 1);   // next tile for this wave
        v8f acc0, acc1;
        #pragma unroll
        for (int r = 0; r < 8; ++r) { acc0[r] = 0.f; acc1[r] = 0.f; }
        #pragma unroll
        for (int c = 0; c < CHUNKS; ++c) {
            v16bf B = load_bf16_frag(bb + c * 32);
            acc0 = __builtin_amdgcn_wmma_f32_16x16x32_bf16(false, A0[c], false, B,
                                                           (short)0, acc0, false, false);
            acc1 = __builtin_amdgcn_wmma_f32_16x16x32_bf16(false, A1[c], false, B,
                                                           (short)0, acc1, false, false);
        }
        const float nx = norm_x[j0 + hl];
        const int   tg = targets[j0 + hl];
        #pragma unroll
        for (int r = 0; r < 16; ++r) {
            const float dot   = (r < 8) ? acc0[r] : acc1[r - 8];
            const float d2    = nc[r] + nx - 2.0f * dot;
            const float dist  = __builtin_amdgcn_sqrtf(fmaxf(d2, 1e-12f));
            const float isval = (dist > EPS) ? 1.0f : 0.0f;
            const float isneg = (cid[r] != tg) ? 1.0f : 0.0f;
            const float wneg  = isneg * isval;
            const float wpos  = isval - wneg;
            negsum[r] = fmaf(wneg, dist, negsum[r]);
            negcnt[r] += wneg;
            possum = fmaf(wpos, dist, possum);
            poscnt += wpos;
        }
    }

    #pragma unroll
    for (int r = 0; r < 16; ++r)
        #pragma unroll
        for (int m = 1; m < 16; m <<= 1) {       // reduce over the 16 columns per half
            negsum[r] += __shfl_xor(negsum[r], m, 32);
            negcnt[r] += __shfl_xor(negcnt[r], m, 32);
        }
    #pragma unroll
    for (int m = 1; m < 32; m <<= 1) {
        possum += __shfl_xor(possum, m, 32);
        poscnt += __shfl_xor(poscnt, m, 32);
    }
    if (hl == 0) {
        #pragma unroll
        for (int r = 0; r < 16; ++r) {
            const int lrow = (r < 8) ? (half * 8 + r) : (16 + half * 8 + (r - 8));
            s_sum[wave][lrow] = negsum[r];
            s_cnt[wave][lrow] = negcnt[r];
        }
    }
    if (lane == 0) { s_pos[wave][0] = possum; s_pos[wave][1] = poscnt; }
    __syncthreads();
    if (tid < 32) {
        float s = 0.f, c = 0.f;
        #pragma unroll
        for (int w = 0; w < 8; ++w) { s += s_sum[w][tid]; c += s_cnt[w][tid]; }
        s_dneg[tid] = s / c;                      // per-row mean negative distance
    }
    if (tid == 0) {
        float ps = 0.f, pc = 0.f;
        #pragma unroll
        for (int w = 0; w < 8; ++w) { ps += s_pos[w][0]; pc += s_pos[w][1]; }
        pos_part[blockIdx.x * 2 + 0] = ps;
        pos_part[blockIdx.x * 2 + 1] = pc;
    }
    __syncthreads();
    float dneg[16];
    #pragma unroll
    for (int r = 0; r < 16; ++r) {
        const int lrow = (r < 8) ? (half * 8 + r) : (16 + half * 8 + (r - 8));
        dneg[r] = s_dneg[lrow];
    }

    // ---------------- pass 2: hard negatives (dist < per-row mean) --------------
    float hsum[16], hcnt[16];
    #pragma unroll
    for (int r = 0; r < 16; ++r) { hsum[r] = 0.f; hcnt[r] = 0.f; }

    for (int t = wave; t < ntiles; t += 8) {
        const int j0 = t << 4;
        const unsigned short* bb = inputs_bf + (size_t)(j0 + hl) * DIMS + half * 8;
        __builtin_prefetch(bb + (size_t)128 * DIMS, 0, 1);
        v8f acc0, acc1;
        #pragma unroll
        for (int r = 0; r < 8; ++r) { acc0[r] = 0.f; acc1[r] = 0.f; }
        #pragma unroll
        for (int c = 0; c < CHUNKS; ++c) {
            v16bf B = load_bf16_frag(bb + c * 32);
            acc0 = __builtin_amdgcn_wmma_f32_16x16x32_bf16(false, A0[c], false, B,
                                                           (short)0, acc0, false, false);
            acc1 = __builtin_amdgcn_wmma_f32_16x16x32_bf16(false, A1[c], false, B,
                                                           (short)0, acc1, false, false);
        }
        const float nx = norm_x[j0 + hl];
        const int   tg = targets[j0 + hl];
        #pragma unroll
        for (int r = 0; r < 16; ++r) {
            const float dot  = (r < 8) ? acc0[r] : acc1[r - 8];
            const float d2   = nc[r] + nx - 2.0f * dot;
            const float dist = __builtin_amdgcn_sqrtf(fmaxf(d2, 1e-12f));
            const float w = ((cid[r] != tg) & (dist > EPS) & (dist < dneg[r])) ? 1.0f : 0.0f;
            hsum[r] = fmaf(w, dist, hsum[r]);
            hcnt[r] += w;
        }
    }

    #pragma unroll
    for (int r = 0; r < 16; ++r)
        #pragma unroll
        for (int m = 1; m < 16; m <<= 1) {
            hsum[r] += __shfl_xor(hsum[r], m, 32);
            hcnt[r] += __shfl_xor(hcnt[r], m, 32);
        }
    __syncthreads();   // s_sum/s_cnt pass-1 consumption finished above
    if (hl == 0) {
        #pragma unroll
        for (int r = 0; r < 16; ++r) {
            const int lrow = (r < 8) ? (half * 8 + r) : (16 + half * 8 + (r - 8));
            s_sum[wave][lrow] = hsum[r];
            s_cnt[wave][lrow] = hcnt[r];
        }
    }
    __syncthreads();
    if (tid < 32) {
        float s = 0.f, c = 0.f;
        #pragma unroll
        for (int w = 0; w < 8; ++w) { s += s_sum[w][tid]; c += s_cnt[w][tid]; }
        row_an[i0 + tid] = s / c;                 // per-center hard-negative mean
    }
}

// ---------------- Phase C: deterministic final scalar ----------------------------
__global__ __launch_bounds__(256) void dcl_finalize(const float* __restrict__ row_an,
                                                    const float* __restrict__ pos_part,
                                                    float* __restrict__ out,
                                                    int id_num, int nblocks) {
    __shared__ float red[256];
    float a = 0.f;
    for (int i = threadIdx.x; i < id_num; i += 256) a += row_an[i];
    red[threadIdx.x] = a;
    __syncthreads();
    for (int off = 128; off; off >>= 1) {
        if (threadIdx.x < off) red[threadIdx.x] += red[threadIdx.x + off];
        __syncthreads();
    }
    if (threadIdx.x == 0) {
        float ps = 0.f, pc = 0.f;
        for (int b = 0; b < nblocks; ++b) { ps += pos_part[2 * b]; pc += pos_part[2 * b + 1]; }
        const float ap_mean = ps / pc;
        const float an_mean = red[0] / (float)id_num;
        out[0] = ap_mean / an_mean;
    }
}

extern "C" void kernel_launch(void* const* d_in, const int* in_sizes, int n_in,
                              void* d_out, int out_size, void* d_ws, size_t ws_size,
                              hipStream_t stream) {
    const float* inputs  = (const float*)d_in[0];
    const int*   targets = (const int*)d_in[1];
    const int N       = in_sizes[0] / DIMS;   // 32768
    const int id_num  = N / 8;                // 4096 (N / TEMPS / NUM_POS)
    const int mblocks = id_num / 32;          // 128

    char* ws = (char*)d_ws;
    unsigned short* inputs_bf  = (unsigned short*)ws; ws += (size_t)N * DIMS * 2;
    unsigned short* centers_bf = (unsigned short*)ws; ws += (size_t)id_num * DIMS * 2;
    float* norm_x   = (float*)ws; ws += (size_t)N * 4;
    float* norm_c   = (float*)ws; ws += (size_t)id_num * 4;
    float* row_an   = (float*)ws; ws += (size_t)id_num * 4;
    float* pos_part = (float*)ws;

    dcl_centers <<<id_num, 256, 0, stream>>>(inputs, centers_bf, norm_c, N);
    dcl_rows    <<<N,      256, 0, stream>>>(inputs, inputs_bf, norm_x);
    dcl_main    <<<mblocks, 256, 0, stream>>>(inputs_bf, centers_bf, norm_x, norm_c,
                                              targets, row_an, pos_part, N);
    dcl_finalize<<<1,      256, 0, stream>>>(row_an, pos_part, (float*)d_out, id_num, mblocks);
}
